// RNN_63934883168911
// MI455X (gfx1250) — matrix operand order repaired and checked by reading, not compiled, not run on hardware
//
#include <hip/hip_runtime.h>
#include <hip/hip_bf16.h>

// ---------------------------------------------------------------------------
// MI455X (gfx1250) RNN LM forward.
//   V=32000, D=100, B=32, T=256.  Output: log_softmax(logits) as (T*B, V) f32.
//
// Roofline: output store = 1.05 GB @ 23.3 TB/s ~= 45us lower bound; emb
// (12.8MB) is L2-resident (192MB L2), so the vocab GEMM is recomputed in a
// second phase (online logsumexp in phase 1) instead of spilling logits.
// Recurrence and vocab matmuls use v_wmma_f32_16x16x32_bf16; U uses f32 fmaf.
// ---------------------------------------------------------------------------

typedef __attribute__((ext_vector_type(16))) __bf16 v16bf;
typedef __attribute__((ext_vector_type(8)))  float  v8f;

#define VOCAB   32000
#define DDIM    100
#define DPAD    128          // K padded to 4 x 32 for bf16 WMMA
#define BATCH   32
#define TSTEPS  256
#define ROWS    (TSTEPS * BATCH)   // 8192
#define VTILES  (VOCAB / 16)       // 2000
#define NW      8                  // waves per block in the big GEMM

union FragU { uint4 q[2]; v16bf v; };

// A-matrix 16x32 bf16 fragment (16b A layout, ISA 7.12.2):
//   elems 0..7  = k = off0 .. off0+7,  elems 8..15 = k = off0+16 .. off0+23
//   with off0 = kt*32 + 8*half ; row = lane%16.
__device__ __forceinline__ v16bf load_frag_A(const __bf16* row, int off0) {
  FragU f;
  f.q[0] = *reinterpret_cast<const uint4*>(row + off0);
  f.q[1] = *reinterpret_cast<const uint4*>(row + off0 + 16);
  return f.v;
}

// B-matrix 32x16 bf16 fragment: lane n (col) holds k = off0..off0+7 and
// off0+16..off0+23 with off0 = kt*32 + 8*half (same order as A).
__device__ __forceinline__ v16bf load_frag_B(const __bf16* row, int off0) {
  FragU f;
  f.q[0] = *reinterpret_cast<const uint4*>(row + off0);
  f.q[1] = *reinterpret_cast<const uint4*>(row + off0 + 16);
  return f.v;
}

__device__ __forceinline__ v8f wmma_bf16(v16bf a, v16bf b, v8f c) {
  return __builtin_amdgcn_wmma_f32_16x16x32_bf16(
      false, a, false, b, (short)0, c, false, false);
}

// Branch-free tanh.  gfx1250 has V_TANH_F32 (TRANS op); fall back to a
// v_exp_f32 + v_rcp_f32 formulation with no EXEC-mask branches.
__device__ __forceinline__ float fast_tanh(float x) {
#if __has_builtin(__builtin_amdgcn_tanhf)
  return __builtin_amdgcn_tanhf(x);
#else
  float ax = fabsf(x);
  float t  = __expf(-2.0f * ax);                        // in (0,1]
  float r  = (1.0f - t) * __builtin_amdgcn_rcpf(1.0f + t);
  return copysignf(r, x);
#endif
}

// ---------------------------------------------------------------------------
// Kernel 1: emb (32000x100 f32) -> Eb (32000x128 bf16, zero-padded K)
// ---------------------------------------------------------------------------
__global__ void __launch_bounds__(256)
k_convert_emb(const float* __restrict__ emb, __bf16* __restrict__ Eb) {
  int idx = blockIdx.x * 256 + threadIdx.x;      // over 32000*128
  int v = idx >> 7, d = idx & 127;
  float f = (d < DDIM) ? emb[v * DDIM + d] : 0.0f;
  Eb[idx] = (__bf16)f;
}

// ---------------------------------------------------------------------------
// Kernel 2: U[t*32+b, j] = b1[j] + sum_d emb[tok[b,t], d] * W1[j, d]
// (the input-dependent half of GEMM1 has no serial dependence -> parallel)
// ---------------------------------------------------------------------------
__global__ void __launch_bounds__(128)
k_compute_U(const int* __restrict__ tok, const float* __restrict__ emb,
            const float* __restrict__ W1, const float* __restrict__ b1,
            float* __restrict__ U) {
  __shared__ float xs[DDIM];
  const int row = blockIdx.x;                     // t*32 + b
  const int t = row >> 5, b = row & 31;
  const int token = tok[b * TSTEPS + t];
  const int j = threadIdx.x;
  if (j < DDIM) xs[j] = emb[token * DDIM + j];
  __syncthreads();
  if (j < DDIM) {
    const float* w = W1 + j * 200;                // x-part = W1[j, 0..99]
    float acc = b1[j];
#pragma unroll 4
    for (int d = 0; d < DDIM; ++d) acc = fmaf(xs[d], w[d], acc);
    U[row * DDIM + j] = acc;
  }
}

// ---------------------------------------------------------------------------
// Kernel 3: serial recurrence, one workgroup, 14 waves = 2 Mtiles x 7 Ntiles.
// Weight B-fragments (bf16) live in VGPRs across the whole time loop; h state
// is double-buffered in LDS (K-padded to 128).  h2[t] goes straight to Hb
// (bf16, row = t*32 + b), with the global store deferred past the closing
// barrier so its completion wait overlaps the next step's GEMM1.
// ---------------------------------------------------------------------------
__global__ void __launch_bounds__(448)
k_recurrence(const float* __restrict__ W1, const float* __restrict__ W2,
             const float* __restrict__ b2, const float* __restrict__ U,
             __bf16* __restrict__ Hb) {
  __shared__ __align__(16) __bf16 h1buf[2][BATCH][DPAD];
  __shared__ __align__(16) __bf16 h2buf[2][BATCH][DPAD];

  const int wave = threadIdx.x >> 5;   // 0..13
  const int lane = threadIdx.x & 31;
  const int mt   = wave / 7;           // row tile (batch 0..15 / 16..31)
  const int nt   = wave % 7;           // col tile (j)
  const int n    = lane & 15;
  const int half = lane >> 4;
  const int j    = nt * 16 + n;        // output hidden index (valid if <100)
  const bool valid = (j < DDIM);
  const int  jc    = valid ? j : (DDIM - 1);  // clamped, always in-bounds

  // ---- build weight B-fragments in registers (held across the time loop) --
  v16bf w1B[4];                        // W1h^T  (K padded to 128)
#pragma unroll
  for (int kt = 0; kt < 4; ++kt) {
    FragU f;
#pragma unroll
    for (int e = 0; e < 16; ++e) {
      int k = kt * 32 + 8 * half + (e < 8 ? e : e + 8);
      float v = (k < DDIM && valid) ? W1[j * 200 + 100 + k] : 0.0f;
      f.v[e] = (__bf16)v;
    }
    w1B[kt] = f.v;
  }
  v16bf w2B[8];                        // [h1-part | h2-part], each K-padded 128
#pragma unroll
  for (int kt = 0; kt < 8; ++kt) {
    FragU f;
#pragma unroll
    for (int e = 0; e < 16; ++e) {
      int kk = kt * 32 + 8 * half + (e < 8 ? e : e + 8);     // 0..255 padded cat-K space
      float v = 0.0f;
      if (valid) {
        if (kk < DPAD) { if (kk < DDIM) v = W2[j * 200 + kk]; }
        else { int k2 = kk - DPAD; if (k2 < DDIM) v = W2[j * 200 + 100 + k2]; }
      }
      f.v[e] = (__bf16)v;
    }
    w2B[kt] = f.v;
  }
  const float b2v = valid ? b2[j] : 0.0f;

  // ---- zero initial state (h1_0 = h2_0 = 0, padding = 0) -------------------
  for (int idx = threadIdx.x; idx < 2 * BATCH * DPAD; idx += 448) {
    (&h1buf[0][0][0])[idx] = (__bf16)0.0f;
    (&h2buf[0][0][0])[idx] = (__bf16)0.0f;
  }
  __syncthreads();

  const int arow = mt * 16 + (lane & 15);   // A-matrix row (batch index)

  for (int t = 0; t < TSTEPS; ++t) {
    const int par = t & 1, prv = par ^ 1;

    // Hoisted, branchless U loads (overlap global latency with the WMMAs).
    float uv[8];
#pragma unroll
    for (int i = 0; i < 8; ++i) {
      const int M = mt * 16 + i + 8 * half;
      uv[i] = U[(t * BATCH + M) * DDIM + jc];
    }

    // GEMM1: h1_new = tanh(U[t] + h1_prev @ W1h^T)
    {
      const __bf16* ar = &h1buf[prv][arow][0];
      v16bf a[4];
#pragma unroll
      for (int kt = 0; kt < 4; ++kt) a[kt] = load_frag_A(ar, kt * 32 + 8 * half);
      v8f c = {};
#pragma unroll
      for (int kt = 0; kt < 4; ++kt) c = wmma_bf16(a[kt], w1B[kt], c);
#pragma unroll
      for (int i = 0; i < 8; ++i) {
        const int M = mt * 16 + i + 8 * half;          // batch row 0..31
        float h = valid ? fast_tanh(c[i] + uv[i]) : 0.0f;
        h1buf[par][M][j] = (__bf16)h;
      }
    }
    __syncthreads();

    // GEMM2: h2_new = tanh([h1_new | h2_prev] @ W2^T + b2)
    __bf16 hbout[8];
    {
      const __bf16* a1 = &h1buf[par][arow][0];
      const __bf16* a2 = &h2buf[prv][arow][0];
      v16bf a[8];
#pragma unroll
      for (int kt = 0; kt < 4; ++kt) a[kt]     = load_frag_A(a1, kt * 32 + 8 * half);
#pragma unroll
      for (int kt = 0; kt < 4; ++kt) a[4 + kt] = load_frag_A(a2, kt * 32 + 8 * half);
      v8f c2 = {};
#pragma unroll
      for (int kt = 0; kt < 8; ++kt) c2 = wmma_bf16(a[kt], w2B[kt], c2);
#pragma unroll
      for (int i = 0; i < 8; ++i) {
        const int M = mt * 16 + i + 8 * half;
        float h = valid ? fast_tanh(c2[i] + b2v) : 0.0f;
        __bf16 hb = (__bf16)h;
        h2buf[par][M][j] = hb;
        hbout[i] = hb;
      }
    }
    __syncthreads();

    // Deferred global store of h2[t] (after the barrier): its storecnt wait
    // lands at the NEXT step's mid barrier, hidden behind GEMM1.
#pragma unroll
    for (int i = 0; i < 8; ++i) {
      const int M = mt * 16 + i + 8 * half;
      Hb[(size_t)(t * BATCH + M) * DPAD + j] = hbout[i];
    }
  }
}

// ---------------------------------------------------------------------------
// Kernel 4: fused logits = Hb @ Eb^T  with log_softmax.
// Each block owns 16 rows x all 32000 cols; 8 waves split the 2000 col-tiles.
// Phase 1: online (max, sum-exp).  Phase 2: recompute tiles (emb is L2-hot)
// and store c - logsumexp.  Output stream (1.05 GB) is the HBM bound.
// ---------------------------------------------------------------------------
__global__ void __launch_bounds__(256)
k_logits_lsm(const __bf16* __restrict__ Hb, const __bf16* __restrict__ Eb,
             float* __restrict__ out) {
  const int lane = threadIdx.x & 31;
  const int wave = threadIdx.x >> 5;           // 0..7
  const int n    = lane & 15;
  const int half = lane >> 4;
  const int rbase = blockIdx.x * 16;

  // A fragments for this block's 16 rows (reused for all 2000 col tiles)
  v16bf A[4];
  {
    const __bf16* ar = Hb + (size_t)(rbase + n) * DPAD;
#pragma unroll
    for (int kt = 0; kt < 4; ++kt) A[kt] = load_frag_A(ar, kt * 32 + 8 * half);
  }

  auto tile = [&](int vt) -> v8f {
    const __bf16* br = Eb + (size_t)(vt * 16 + n) * DPAD + 8 * half;
    v16bf B[4];
#pragma unroll
    for (int kt = 0; kt < 4; ++kt) B[kt] = load_frag_B(br, kt * 32);
    v8f c = {};
#pragma unroll
    for (int kt = 0; kt < 4; ++kt) c = wmma_bf16(A[kt], B[kt], c);
    return c;
  };

  // ---- phase 1: online row max / sum-exp ----------------------------------
  float m[8], s[8];
#pragma unroll
  for (int i = 0; i < 8; ++i) { m[i] = -3.0e38f; s[i] = 0.0f; }

  for (int vt = wave; vt < VTILES; vt += NW) {
    if (vt + NW < VTILES)
      __builtin_prefetch(Eb + (size_t)((vt + NW) * 16 + n) * DPAD + 16 * half, 0, 0);
    v8f c = tile(vt);
#pragma unroll
    for (int i = 0; i < 8; ++i) {
      float cv = c[i];
      float mn = fmaxf(m[i], cv);
      s[i] = s[i] * __expf(m[i] - mn) + __expf(cv - mn);
      m[i] = mn;
    }
  }

  // reduce across the 16 lanes that share each row (xor<=8 stays in group)
#pragma unroll
  for (int i = 0; i < 8; ++i) {
#pragma unroll
    for (int off = 1; off < 16; off <<= 1) {
      float mo = __shfl_xor(m[i], off, 32);
      float so = __shfl_xor(s[i], off, 32);
      float mn = fmaxf(m[i], mo);
      s[i] = s[i] * __expf(m[i] - mn) + so * __expf(mo - mn);
      m[i] = mn;
    }
  }

  __shared__ float Ms[NW][16], Ss[NW][16], L[16];
  if (n == 0) {
#pragma unroll
    for (int i = 0; i < 8; ++i) {
      Ms[wave][i + 8 * half] = m[i];
      Ss[wave][i + 8 * half] = s[i];
    }
  }
  __syncthreads();
  if (threadIdx.x < 16) {
    float M = -3.0e38f;
    for (int w = 0; w < NW; ++w) M = fmaxf(M, Ms[w][threadIdx.x]);
    float S = 0.0f;
    for (int w = 0; w < NW; ++w) S += Ss[w][threadIdx.x] * __expf(Ms[w][threadIdx.x] - M);
    L[threadIdx.x] = M + __logf(S);               // logsumexp per row
  }
  __syncthreads();

  float Lv[8];
#pragma unroll
  for (int i = 0; i < 8; ++i) Lv[i] = L[i + 8 * half];

  // ---- phase 2: recompute tiles, store log_softmax ------------------------
  for (int vt = wave; vt < VTILES; vt += NW) {
    v8f c = tile(vt);
#pragma unroll
    for (int i = 0; i < 8; ++i)
      out[(size_t)(rbase + i + 8 * half) * VOCAB + vt * 16 + n] = c[i] - Lv[i];
  }
}

// ---------------------------------------------------------------------------
extern "C" void kernel_launch(void* const* d_in, const int* in_sizes, int n_in,
                              void* d_out, int out_size, void* d_ws, size_t ws_size,
                              hipStream_t stream) {
  const int*   tok = (const int*)d_in[0];     // (B, T)
  const float* emb = (const float*)d_in[1];   // (V, 100)
  const float* W1  = (const float*)d_in[2];   // (100, 200)
  const float* b1  = (const float*)d_in[3];   // (100,)
  const float* W2  = (const float*)d_in[4];   // (100, 200)
  const float* b2  = (const float*)d_in[5];   // (100,)
  float* out = (float*)d_out;                 // (T*B, V)

  char* ws = (char*)d_ws;
  const size_t EB_BYTES = (size_t)VOCAB * DPAD * 2;   // 8,192,000
  const size_t HB_BYTES = (size_t)ROWS * DPAD * 2;    // 2,097,152
  __bf16* Eb = (__bf16*)ws;
  __bf16* Hb = (__bf16*)(ws + EB_BYTES);
  float*  U  = (float*)(ws + EB_BYTES + HB_BYTES);    // 8192*100*4

  hipMemsetAsync(Hb, 0, HB_BYTES, stream);            // pad cols 112..127

  k_convert_emb<<<(VOCAB * DPAD) / 256, 256, 0, stream>>>(emb, Eb);
  k_compute_U  <<<ROWS, 128, 0, stream>>>(tok, emb, W1, b1, U);
  k_recurrence <<<1, 448, 0, stream>>>(W1, W2, b2, U, Hb);
  k_logits_lsm <<<512, 256, 0, stream>>>(Hb, Eb, out);
}
